// MHFSpectralConv1D_19267223290314
// MI455X (gfx1250) — matrix operand order repaired and checked by reading, not compiled, not run on hardware
//
#include <hip/hip_runtime.h>

// ---------------------------------------------------------------------------
// MHFSpectralConv1D on gfx1250 (MI455X)
//   x:      [B=32, Cin=128, S=8192] f32
//   w_real: [H=4, Cin=128, Hc=32, M=64] f32   (w_imag same)
//   out:    [B, Cout=128, S] f32
//
// Only 64 rfft modes survive -> replace FFTs with two tall-skinny GEMMs
// against cos/sin basis matrices, executed with v_wmma_f32_16x16x32_f16.
// B fragments are batch-loaded (8 per k-step) into distinct registers so a
// single long-latency wait covers 8 WMMAs.
// ---------------------------------------------------------------------------

typedef __attribute__((ext_vector_type(16))) _Float16 v16h;
typedef __attribute__((ext_vector_type(8)))  float    v8f;

#define S_LEN   8192
#define SMASK   8191
#define NB      32
#define CIN     128
#define COUT    128
#define NMODE   64
#define NCOL    128               // 2*NMODE (re | im columns)
#define NROWS   4096              // NB*CIN == NB*COUT
#define KSPLIT  16                // K-partition of the forward DFT
#define TWO_PI_OVER_S 7.6699039394282066e-4f

// ---------------------------------------------------------------------------
// Twiddle tables, stored pre-swizzled into WMMA B-fragment order:
//   fragment element j (0..15) of lane L = element(K = 16*(L/16) + j,
//                                                  N = 16*ntile + (L&15))
// 32 contiguous bytes per lane -> one vector load per fragment in the GEMM.
// ---------------------------------------------------------------------------

// Forward basis E[S, 128]: col n<64 -> cos(2*pi*n*s/S), col 64+m -> -sin(...)
// layout index = ((kt*8 + nt)*32 + lane)*16 + j   (kt = 32-wide s-tile)
__global__ void __launch_bounds__(256) build_fwd_table(_Float16* __restrict__ t1) {
    int idx  = blockIdx.x * 256 + threadIdx.x;   // 0 .. 8192*128-1
    int j    = idx & 15;
    int lane = (idx >> 4) & 31;
    int nt   = (idx >> 9) & 7;
    int kt   = idx >> 12;
    int half = lane >> 4;
    int n    = nt * 16 + (lane & 15);
    int k    = kt * 32 + half * 16 + j;          // sample index s
    int m    = (n < NMODE) ? n : (n - NMODE);
    int r    = (m * k) & SMASK;                  // exact periodic reduction
    float ang = TWO_PI_OVER_S * (float)r;
    float v   = (n < NMODE) ? __cosf(ang) : -__sinf(ang);
    t1[idx] = (_Float16)v;
}

// Inverse basis E2[128, S]: row k<64 -> cos(2*pi*k*s/S), row 64+m -> -sin(...)
// layout index = (((cg*4 + kt)*8 + nt)*32 + lane)*16 + j
//   cg = 128-wide sample group (64), nt = 16-wide tile in group (8),
//   kt = 32-wide K tile (4).  The 8 fragments per (cg,kt) are contiguous.
__global__ void __launch_bounds__(256) build_inv_table(_Float16* __restrict__ t2) {
    int idx  = blockIdx.x * 256 + threadIdx.x;   // 0 .. 128*8192-1
    int j    = idx & 15;
    int lane = (idx >> 4) & 31;
    int nt   = (idx >> 9) & 7;
    int kt   = (idx >> 12) & 3;
    int cg   = idx >> 14;                        // 0..63
    int half = lane >> 4;
    int s    = (cg * 8 + nt) * 16 + (lane & 15);
    int k    = kt * 32 + half * 16 + j;          // 0..127 (mode re/im row)
    int m    = (k < NMODE) ? k : (k - NMODE);
    int r    = (m * s) & SMASK;
    float ang = TWO_PI_OVER_S * (float)r;
    float v   = (k < NMODE) ? __cosf(ang) : -__sinf(ang);
    t2[idx] = (_Float16)v;
}

// Build an A-fragment (contiguous k = 16*half + j layout) from 16 contiguous f32.
__device__ __forceinline__ v16h load_cvt_afrag(const float* __restrict__ base,
                                               int k0, int half) {
    const float4* p0 = (const float4*)(base + k0 + half * 16);
    const float4* p1 = (const float4*)(base + k0 + half * 16 + 8);
    float4 u0 = p0[0], u1 = p0[1];
    float4 w0 = p1[0], w1 = p1[1];
    v16h a;
    a[0]  = (_Float16)u0.x; a[1]  = (_Float16)u0.y;
    a[2]  = (_Float16)u0.z; a[3]  = (_Float16)u0.w;
    a[4]  = (_Float16)u1.x; a[5]  = (_Float16)u1.y;
    a[6]  = (_Float16)u1.z; a[7]  = (_Float16)u1.w;
    a[8]  = (_Float16)w0.x; a[9]  = (_Float16)w0.y;
    a[10] = (_Float16)w0.z; a[11] = (_Float16)w0.w;
    a[12] = (_Float16)w1.x; a[13] = (_Float16)w1.y;
    a[14] = (_Float16)w1.z; a[15] = (_Float16)w1.w;
    return a;
}

// ---------------------------------------------------------------------------
// Stage 1: partial forward DFT.  Xpart[ks][4096, 128] = x_strip @ E_strip
// Each wave: one 16-row strip x all 128 output columns, 1/16th of K.
// 4096 waves total.
// ---------------------------------------------------------------------------
__global__ void __launch_bounds__(256) dft_fwd(const float* __restrict__ x,
                                               const _Float16* __restrict__ t1,
                                               float* __restrict__ Xpart) {
    int wave    = threadIdx.x >> 5;
    int lane    = threadIdx.x & 31;
    int gw      = blockIdx.x * 8 + wave;         // 0..4095
    int ks      = gw & (KSPLIT - 1);
    int rowTile = gw >> 4;                       // 0..255
    int half    = lane >> 4;
    int lane16  = lane & 15;
    const float* xrow = x + (size_t)(rowTile * 16 + lane16) * S_LEN;

    v8f acc[8] = {};
    int kt0 = ks * 16;
    for (int kt = kt0; kt < kt0 + 16; ++kt) {
        const _Float16* tb = t1 + (size_t)kt * 4096 + (size_t)lane * 16;
        // batch all 8 B fragments into distinct registers -> one latency wait
        v16h b[8];
#pragma unroll
        for (int nt = 0; nt < 4; ++nt)
            b[nt] = *(const v16h*)(tb + (size_t)nt * 512);
        // A load + f32->f16 convert overlaps the in-flight B loads
        v16h a = load_cvt_afrag(xrow, kt * 32, half);
#pragma unroll
        for (int nt = 4; nt < 8; ++nt)
            b[nt] = *(const v16h*)(tb + (size_t)nt * 512);
        // pull the B tile two k-steps ahead toward the WGP cache
        __builtin_prefetch((const void*)(tb + 2 * 4096), 0, 1);
#pragma unroll
        for (int nt = 0; nt < 8; ++nt)
            acc[nt] = __builtin_amdgcn_wmma_f32_16x16x32_f16(
                false, a, false, b[nt], (short)0, acc[nt], false, false);
    }
    float* Xo = Xpart + (size_t)ks * NROWS * NCOL;
#pragma unroll
    for (int nt = 0; nt < 8; ++nt)
#pragma unroll
        for (int r = 0; r < 8; ++r)
            Xo[(size_t)(rowTile * 16 + half * 8 + r) * NCOL + nt * 16 + lane16] =
                acc[nt][r];
}

// Sum the K-split partials: X = sum_ks Xpart[ks]
__global__ void __launch_bounds__(256) reduce_X(const float* __restrict__ Xpart,
                                                float* __restrict__ X) {
    int idx = blockIdx.x * 256 + threadIdx.x;    // 0..524287
    float s = 0.f;
#pragma unroll
    for (int k = 0; k < KSPLIT; ++k)
        s += Xpart[(size_t)k * (NROWS * NCOL) + idx];
    X[idx] = s;
}

// ---------------------------------------------------------------------------
// Stage 2: complex mode mixing + irfft scaling folded into coefficients.
//   cre/cim[b, h*32+o, m] = sum_i X[b,i,m] (*) w[h,i,o,m]
//   C2[row, m]    = scale_m * cre      (scale_0 = 1/S, else 2/S)
//   C2[row, 64+m] = scale_m * cim      (m=0 -> 0: irfft drops im(c0))
// ---------------------------------------------------------------------------
__global__ void __launch_bounds__(256) mode_mix(const float* __restrict__ X,
                                                const float* __restrict__ wr,
                                                const float* __restrict__ wi,
                                                float* __restrict__ C2) {
    int t  = blockIdx.x * 256 + threadIdx.x;     // 0..262143
    int m  = t & 63;
    int co = (t >> 6) & 127;
    int b  = t >> 13;
    int h  = co >> 5;
    int o  = co & 31;
    const float* Xb = X  + (size_t)b * CIN * NCOL + m;
    const float* Wr = wr + ((size_t)(h * CIN) * 32 + o) * 64 + m;
    const float* Wi = wi + ((size_t)(h * CIN) * 32 + o) * 64 + m;
    float cre = 0.f, cim = 0.f;
    for (int i = 0; i < CIN; ++i) {
        float xr = Xb[(size_t)i * NCOL];
        float xi = Xb[(size_t)i * NCOL + 64];
        float a  = Wr[(size_t)i * 2048];         // stride Hc*M = 32*64
        float c  = Wi[(size_t)i * 2048];
        cre = fmaf(xr, a, cre); cre = fmaf(-xi, c, cre);
        cim = fmaf(xr, c, cim); cim = fmaf(xi, a, cim);
    }
    float sc = (m == 0) ? (1.0f / S_LEN) : (2.0f / S_LEN);
    size_t row = (size_t)b * COUT + co;
    C2[row * NCOL + m]      = cre * sc;
    C2[row * NCOL + 64 + m] = (m == 0) ? 0.0f : cim * sc;
}

// ---------------------------------------------------------------------------
// Stage 3: partial inverse DFT.  y[4096, 8192] = C2[4096, 128] @ E2[128, 8192]
// Each wave: 16-row strip x 128 output samples (8 WMMA tiles), K = 128.
// 16384 waves total.
// ---------------------------------------------------------------------------
__global__ void __launch_bounds__(256) dft_inv(const float* __restrict__ C2,
                                               const _Float16* __restrict__ t2,
                                               float* __restrict__ y) {
    int wave    = threadIdx.x >> 5;
    int lane    = threadIdx.x & 31;
    int gw      = blockIdx.x * 8 + wave;         // 0..16383
    int colGrp  = gw & 63;                       // 128 samples each
    int rowTile = gw >> 6;                       // 0..255
    int half    = lane >> 4;
    int lane16  = lane & 15;
    const float* crow = C2 + (size_t)(rowTile * 16 + lane16) * NCOL;

    v8f acc[8] = {};
#pragma unroll
    for (int kt = 0; kt < 4; ++kt) {
        const _Float16* tb = t2 + ((size_t)(colGrp * 4 + kt) * 8) * 512 +
                             (size_t)lane * 16;
        v16h b[8];
#pragma unroll
        for (int nt = 0; nt < 4; ++nt)
            b[nt] = *(const v16h*)(tb + (size_t)nt * 512);
        v16h a = load_cvt_afrag(crow, kt * 32, half);
#pragma unroll
        for (int nt = 4; nt < 8; ++nt)
            b[nt] = *(const v16h*)(tb + (size_t)nt * 512);
#pragma unroll
        for (int nt = 0; nt < 8; ++nt)
            acc[nt] = __builtin_amdgcn_wmma_f32_16x16x32_f16(
                false, a, false, b[nt], (short)0, acc[nt], false, false);
    }
#pragma unroll
    for (int nt = 0; nt < 8; ++nt)
#pragma unroll
        for (int r = 0; r < 8; ++r)
            y[(size_t)(rowTile * 16 + half * 8 + r) * S_LEN +
              (size_t)(colGrp * 8 + nt) * 16 + lane16] = acc[nt][r];
}

// ---------------------------------------------------------------------------
extern "C" void kernel_launch(void* const* d_in, const int* in_sizes, int n_in,
                              void* d_out, int out_size, void* d_ws, size_t ws_size,
                              hipStream_t stream) {
    const float* x      = (const float*)d_in[0];
    const float* w_real = (const float*)d_in[1];
    const float* w_imag = (const float*)d_in[2];
    float* out = (float*)d_out;

    char* ws = (char*)d_ws;
    _Float16* t1    = (_Float16*)(ws);                       //  2 MB fwd basis
    _Float16* t2    = (_Float16*)(ws + (2ull  << 20));       //  2 MB inv basis
    float*    Xpart = (float*)   (ws + (4ull  << 20));       // 32 MB partials
    float*    X     = (float*)   (ws + (36ull << 20));       //  2 MB spectrum
    float*    C2    = (float*)   (ws + (38ull << 20));       //  2 MB coeffs

    build_fwd_table<<<4096, 256, 0, stream>>>(t1);
    build_inv_table<<<4096, 256, 0, stream>>>(t2);
    dft_fwd<<<512, 256, 0, stream>>>(x, t1, Xpart);          // 4096 waves
    reduce_X<<<2048, 256, 0, stream>>>(Xpart, X);
    mode_mix<<<1024, 256, 0, stream>>>(X, w_real, w_imag, C2);
    dft_inv<<<2048, 256, 0, stream>>>(C2, t2, out);          // 16384 waves
}
